// MetaMu2_34437047779850
// MI455X (gfx1250) — matrix operand order repaired and checked by reading, not compiled, not run on hardware
//
#include <hip/hip_runtime.h>
#include <hip/hip_bf16.h>

#define SEQ     512
#define BATCH   64
#define IN_DIM  64
#define HID     512
#define MID     256
#define CONCAT  1088   // IN + 2*HID
#define ZDIM    320    // IN + MID
#define NWG     8
#define NTHREADS 256

#define USE_ASYNC_LDS 1   // gfx1250 global_load_async_to_lds_b128 path

typedef __bf16 bf16_t;
typedef __attribute__((ext_vector_type(4)))  __bf16 v4bf;
typedef __attribute__((ext_vector_type(8)))  __bf16 v8bf;
typedef __attribute__((ext_vector_type(16))) __bf16 v16bf;
typedef __attribute__((ext_vector_type(8)))  float  v8f;

// ---------------- fragment loads (CDNA5 wave32 WMMA layouts) ----------------
// A-matrix (16x32 bf16, MxK). Lane L: M = L&15, half = L>>4.
// VGPR0..3: K = half*8 + 0..7 (pairs), VGPR4..7: K = 16 + half*8 + 0..7.
__device__ __forceinline__ v16bf load_a_frag(const bf16_t* base, int row, int half,
                                             int kbase, int stride) {
    const bf16_t* p0 = base + row * stride + kbase + half * 8;
    v8bf lo = *(const v8bf*)p0;        // 16B aligned -> ds_load_b128
    v8bf hi = *(const v8bf*)(p0 + 16);
    return __builtin_shufflevector(lo, hi, 0,1,2,3,4,5,6,7,8,9,10,11,12,13,14,15);
}
// B-matrix (32x16 bf16, KxN). Lane L: N = L&15, holds K = half*8 + 0..7, 16 + half*8 + 0..7.
__device__ __forceinline__ v16bf load_b_frag(const bf16_t* base, int ncol, int half,
                                             int kbase, int stride) {
    const bf16_t* p0 = base + ncol * stride + kbase + half * 8;
    v8bf lo = *(const v8bf*)p0;
    v8bf hi = *(const v8bf*)(p0 + 16);
    return __builtin_shufflevector(lo, hi, 0,1,2,3,4,5,6,7,8,9,10,11,12,13,14,15);
}
__device__ __forceinline__ v8f wmma_bf16(v16bf a, v16bf b, v8f c) {
    return __builtin_amdgcn_wmma_f32_16x16x32_bf16(false, a, false, b, (short)0, c,
                                                   false, false);
}

// ---------------- branch-free activations (v_exp_f32 + v_rcp_f32) ----------
__device__ __forceinline__ float fast_tanh(float x) {
    float xc = fminf(fmaxf(x, -15.0f), 15.0f);
    float e  = __expf(2.0f * xc);
    return (e - 1.0f) * __builtin_amdgcn_rcpf(e + 1.0f);
}
__device__ __forceinline__ float fast_sigmoid(float x) {
    float xc = fminf(fmaxf(x, -30.0f), 30.0f);
    return __builtin_amdgcn_rcpf(1.0f + __expf(-xc));
}

// ---------------- async global->LDS 16B copy (ASYNCcnt tracked) -------------
__device__ __forceinline__ void copy16_g2l(bf16_t* ldst, const bf16_t* gsrc) {
#if USE_ASYNC_LDS
    unsigned lds_off = (unsigned)(unsigned long long)(void*)ldst;
    asm volatile("global_load_async_to_lds_b128 %0, %1, off"
                 :: "v"(lds_off), "v"(gsrc) : "memory");
#else
    *(uint4*)ldst = *(const uint4*)gsrc;
#endif
}
__device__ __forceinline__ void async_copy_wait() {
#if USE_ASYNC_LDS
    asm volatile("s_wait_asynccnt 0x0" ::: "memory");
#endif
}

// ---------------- grid-wide barrier (monotonic counter) ---------------------
__device__ __forceinline__ void grid_barrier(unsigned* cnt, unsigned target) {
    __syncthreads();
    if (threadIdx.x == 0) {
        __threadfence();
        atomicAdd(cnt, 1u);
        volatile unsigned* vc = (volatile unsigned*)cnt;
        while (*vc < target) __builtin_amdgcn_s_sleep(2);
        __threadfence();
    }
    __syncthreads();
}

__global__ __launch_bounds__(NTHREADS, 1)
void metamu2_persistent_kernel(const float* __restrict__ x,
                               const float* __restrict__ old_m,
                               const float* __restrict__ old_s,
                               const float* __restrict__ Wz, const float* __restrict__ bz,
                               const float* __restrict__ Ws, const float* __restrict__ bs,
                               const float* __restrict__ Wm, const float* __restrict__ bm,
                               float* __restrict__ out,
                               bf16_t* __restrict__ zmid_bf,   // [64][256] bf16
                               bf16_t* __restrict__ panel,     // [64][1024] bf16: m | 1/s
                               unsigned* __restrict__ barrier_cnt) {
    extern __shared__ unsigned char smem_raw[];
    float*  s_bz  = (float*)smem_raw;                  // 32
    float*  s_bs  = s_bz + 32;                         // 64
    float*  s_bm  = s_bs + 64;                         // 64
    bf16_t* s_act = (bf16_t*)(s_bm + 64);              // [64][1088] activations
    bf16_t* s_wz  = s_act + BATCH * CONCAT;            // [32][1088] W_z slice
    bf16_t* s_ws  = s_wz + 32 * CONCAT;                // [64][320]  W_s slice
    bf16_t* s_wm  = s_ws + 64 * ZDIM;                  // [64][320]  W_m slice

    const int tid  = threadIdx.x;
    const int wg   = blockIdx.x;
    const int wv   = tid >> 5;        // 8 wave32 waves
    const int lane = tid & 31;
    const int half = lane >> 4;
    const int l16  = lane & 15;

    // ---- one-time: weight slices -> LDS as bf16 ----
    for (int i = tid; i < 32 * CONCAT; i += NTHREADS) {
        int r = i / CONCAT, k = i % CONCAT;
        s_wz[i] = (bf16_t)Wz[(size_t)(wg * 32 + r) * CONCAT + k];
    }
    for (int i = tid; i < 64 * ZDIM; i += NTHREADS) {
        int r = i / ZDIM, k = i % ZDIM;
        s_ws[i] = (bf16_t)Ws[(size_t)(wg * 64 + r) * ZDIM + k];
        s_wm[i] = (bf16_t)Wm[(size_t)(wg * 64 + r) * ZDIM + k];
    }
    if (tid < 32) s_bz[tid] = bz[wg * 32 + tid];
    if (tid < 64) { s_bs[tid] = bs[wg * 64 + tid]; s_bm[tid] = bm[wg * 64 + tid]; }
    __syncthreads();

    // ---- one-time: cooperative bf16 staging panel from initial state ----
    for (int i = wg * NTHREADS + tid; i < BATCH * 1024; i += NWG * NTHREADS) {
        int b = i >> 10, c = i & 1023;
        float v = (c < HID) ? old_m[(size_t)b * HID + c]
                            : 1.0f / old_s[(size_t)b * HID + (c - HID)];
        panel[i] = (bf16_t)v;
    }
    unsigned epoch = 1;
    grid_barrier(barrier_cnt, epoch * NWG);

    float* outM = out;
    float* outS = out + (size_t)SEQ * BATCH * HID;

    for (int t = 0; t < SEQ; ++t) {
        const float* mP = (t == 0) ? old_m : outM + (size_t)(t - 1) * BATCH * HID;
        const float* sP = (t == 0) ? old_s : outS + (size_t)(t - 1) * BATCH * HID;

        // ---- phase A fill: act[:,64:576] <- staging panel[:,0:512] (async b128 copies) ----
        for (int i = tid; i < BATCH * 64; i += NTHREADS) {       // 64 rows x 64 chunks
            int b = i >> 6, c = i & 63;                          // 16B chunk index
            copy16_g2l(s_act + b * CONCAT + 64 + c * 8, panel + b * 1024 + c * 8);
        }
        // act[:,0:64] <- x_t (fp32 -> bf16 convert, float4 granularity)
        for (int i = tid; i < BATCH * IN_DIM / 4; i += NTHREADS) {
            int b = i >> 4, c4 = (i & 15) * 4;
            float4 f = *(const float4*)(x + (size_t)t * BATCH * IN_DIM + b * IN_DIM + c4);
            v4bf w; w[0] = (bf16_t)f.x; w[1] = (bf16_t)f.y;
                    w[2] = (bf16_t)f.z; w[3] = (bf16_t)f.w;
            *(v4bf*)(s_act + b * CONCAT + c4) = w;
        }
        if (t + 1 < SEQ)   // global_prefetch_b8: warm next step's x in L2
            __builtin_prefetch(&x[(size_t)(t + 1) * BATCH * IN_DIM + tid * 16], 0, 1);
        async_copy_wait();
        __syncthreads();

        // ---- phase A GEMM: z_mid cols [wg*32, wg*32+32), two independent chains ----
        {
            const int mt = wv & 3, nt = wv >> 2;
            v8f acc0 = {}, acc1 = {};
            #pragma unroll 1
            for (int kk = 0; kk < CONCAT / 32; kk += 2) {
                acc0 = wmma_bf16(load_a_frag(s_act, mt * 16 + l16, half, kk * 32, CONCAT),
                                 load_b_frag(s_wz,  nt * 16 + l16, half, kk * 32, CONCAT),
                                 acc0);
                acc1 = wmma_bf16(load_a_frag(s_act, mt * 16 + l16, half, kk * 32 + 32, CONCAT),
                                 load_b_frag(s_wz,  nt * 16 + l16, half, kk * 32 + 32, CONCAT),
                                 acc1);
            }
            v8f acc = acc0 + acc1;
            const int   nloc = nt * 16 + l16;
            const int   ng   = wg * 32 + nloc;
            const float bias = s_bz[nloc];
            #pragma unroll
            for (int j = 0; j < 8; ++j) {
                int b = mt * 16 + half * 8 + j;          // C: VGPR j -> M=j (+8 hi half)
                zmid_bf[b * MID + ng] = (bf16_t)fast_tanh(acc[j] + bias);
            }
        }
        epoch++; grid_barrier(barrier_cnt, epoch * NWG);

        // ---- phase B fill: act[:,64:320] <- z_mid (async b128 copies) ----
        for (int i = tid; i < BATCH * 32; i += NTHREADS) {       // 64 rows x 32 chunks
            int b = i >> 5, c = i & 31;
            copy16_g2l(s_act + b * CONCAT + 64 + c * 8, zmid_bf + b * MID + c * 8);
        }
        async_copy_wait();
        __syncthreads();

        // ---- phase B GEMM: hidden rows [wg*64, wg*64+64) ----
        #pragma unroll
        for (int pp = 0; pp < 2; ++pp) {
            const int p  = wv + pp * 8;
            const int mt = p & 3, nt = p >> 2;
            v8f accS = {}, accM = {};
            #pragma unroll 2
            for (int kk = 0; kk < ZDIM / 32; ++kk) {
                v16bf a  = load_a_frag(s_act, mt * 16 + l16, half, kk * 32, CONCAT);
                v16bf bS = load_b_frag(s_ws,  nt * 16 + l16, half, kk * 32, ZDIM);
                v16bf bM = load_b_frag(s_wm,  nt * 16 + l16, half, kk * 32, ZDIM);
                accS = wmma_bf16(a, bS, accS);
                accM = wmma_bf16(a, bM, accM);
            }
            const int   hloc  = nt * 16 + l16;
            const int   hg    = wg * 64 + hloc;
            const float biasS = s_bs[hloc];
            const float biasM = s_bm[hloc];
            #pragma unroll
            for (int j = 0; j < 8; ++j) {
                int    b    = mt * 16 + half * 8 + j;
                float  sOld = sP[(size_t)b * HID + hg];
                float  mOld = mP[(size_t)b * HID + hg];
                float  sNew = sOld + fast_sigmoid(accS[j] + biasS);
                float  rS   = __builtin_amdgcn_rcpf(sNew);
                float  gate = sOld * rS;
                float  mNew = gate * mOld + (1.0f - gate) * fast_tanh(accM[j] + biasM);
                outM[(size_t)t * BATCH * HID + (size_t)b * HID + hg] = mNew;
                outS[(size_t)t * BATCH * HID + (size_t)b * HID + hg] = sNew;
                // producer-side bf16 staging for next step's activation panel
                panel[b * 1024 + hg]       = (bf16_t)mNew;
                panel[b * 1024 + HID + hg] = (bf16_t)rS;
            }
        }
        epoch++; grid_barrier(barrier_cnt, epoch * NWG);
    }
}

extern "C" void kernel_launch(void* const* d_in, const int* in_sizes, int n_in,
                              void* d_out, int out_size, void* d_ws, size_t ws_size,
                              hipStream_t stream) {
    const float* x     = (const float*)d_in[0];
    const float* old_m = (const float*)d_in[1];
    const float* old_s = (const float*)d_in[2];
    const float* Wz    = (const float*)d_in[3];
    const float* bz    = (const float*)d_in[4];
    const float* Ws    = (const float*)d_in[5];
    const float* bs    = (const float*)d_in[6];
    const float* Wm    = (const float*)d_in[7];
    const float* bm    = (const float*)d_in[8];
    float* out = (float*)d_out;

    unsigned* barrier_cnt = (unsigned*)d_ws;                       // 256 B
    bf16_t*   zmid_bf     = (bf16_t*)((char*)d_ws + 256);          // 32 KB
    bf16_t*   panel       = (bf16_t*)((char*)d_ws + 256 + 32768);  // 128 KB

    hipMemsetAsync(d_ws, 0, 256, stream);   // barrier counter := 0 (capture-safe)

    size_t smem = 160 * sizeof(float) +
                  (size_t)(BATCH * CONCAT + 32 * CONCAT + 2 * 64 * ZDIM) * sizeof(bf16_t);
    // ~285 KB of the WGP's 320 KB LDS

    metamu2_persistent_kernel<<<NWG, NTHREADS, smem, stream>>>(
        x, old_m, old_s, Wz, bz, Ws, bs, Wm, bm, out, zmid_bf, panel, barrier_cnt);
}
